// BreakthroughSNN_14740327760332
// MI455X (gfx1250) — matrix operand order repaired and checked by reading, not compiled, not run on hardware
//
#include <hip/hip_runtime.h>

// ---------------------------------------------------------------------------
// BreakthroughSNN for MI455X (gfx1250, wave32, WMMA).
//
//   1) prep_kernel : weights f32 -> bf16, transposed N-major (contiguous
//                    b128 WMMA B-fragments); zero states + barrier counter.
//   2) enc_gemm    : gathered [2048,1024]x[1024,1024] bf16 WMMA, 2 N-tiles
//                    per wave to amortize f32->bf16 A conversion.
//   3) scan_kernel : persistent 16-WGP sequential scan; WMMA GEMMs from
//                    LDS-staged layernormed activations; inference GEMM
//                    K-split across 4 waves + LDS reduce (halves the
//                    critical WMMA chain in the latency-bound phase).
//   4) out_gemm    : [2048,512]x[512,32000] bf16 WMMA, 64x256 block tile,
//                    8 N-tiles/wave sharing one A fragment (store-bound at
//                    23.3 TB/s; ~1 GB L2 B-traffic, 262 MB f32 logits out).
// ---------------------------------------------------------------------------

typedef __bf16 bf16;
typedef __attribute__((ext_vector_type(16))) __bf16 v16bf;
typedef __attribute__((ext_vector_type(8)))  __bf16 v8bf;
typedef __attribute__((ext_vector_type(8)))  float  v8f;

union BF16x16 { v16bf v; v8bf h[2]; };

#define WMMA_BF16(a, b, c) \
  __builtin_amdgcn_wmma_f32_16x16x32_bf16(false, (a), false, (b), (short)0, (c), false, false)

static constexpr int   Bsz     = 16;
static constexpr int   Sseq    = 128;
static constexpr int   Vocab   = 32000;
static constexpr int   DM      = 1024;   // D_MODEL
static constexpr int   DS      = 512;    // D_STATE
static constexpr int   Lnum    = 2;
static constexpr int   Tnum    = 4;
static constexpr int   NTOK    = Bsz * Sseq;          // 2048
static constexpr float DECAYF  = 0.6065306597126334f; // exp(-1/2)
static constexpr float THRF    = 1.0f;
static constexpr int   NBLK    = 16;                  // scan workgroups

// ---------------- workspace layout (bytes, 256-aligned) --------------------
static constexpr size_t OFF_WENCT = 0;                                   // [DM][DM] bf16
static constexpr size_t OFF_WGT   = OFF_WENCT + (size_t)DM*DM*2;         // [L][DM][DS] bf16 (N-major)
static constexpr size_t OFF_WIT   = OFF_WGT   + (size_t)Lnum*DM*DS*2;    // [L][DS][DM] bf16 (N-major)
static constexpr size_t OFF_WOUTT = OFF_WIT   + (size_t)Lnum*DS*DM*2;    // [V][DS] bf16 (N-major)
static constexpr size_t OFF_ENCP  = OFF_WOUTT + (size_t)Vocab*DS*2;      // [2048][DM] f32
static constexpr size_t OFF_HS    = OFF_ENCP  + (size_t)NTOK*DM*4;       // [2048][DS] bf16
static constexpr size_t OFF_ST    = OFF_HS    + (size_t)NTOK*DS*2;       // [L][16][DS] f32
static constexpr size_t OFF_ERR   = OFF_ST    + (size_t)Lnum*Bsz*DS*4;   // [16][DM] f32
static constexpr size_t OFF_BAR   = OFF_ERR   + (size_t)Bsz*DM*4;        // unsigned[64]

// ---------------------------------------------------------------------------
// Fragment loaders per CDNA5 ISA 7.12.2 (16-bit A 16x32 / B 32x16 layouts).
// ---------------------------------------------------------------------------
__device__ inline v16bf load_a_frag(const bf16* A, int ld, int k0, int lane) {
  // A row-major [16, K]: lane row = lane&15, group g = lane>>4.
  // slots 0..7  -> K = k0 + 8g .. +7 ; slots 8..15 -> K = k0 + 16 + 8g .. +7
  const int m = lane & 15, g = lane >> 4;
  const bf16* base = A + (long)m * ld + k0;
  BF16x16 a;
  a.h[0] = *(const v8bf*)(base + 8 * g);
  a.h[1] = *(const v8bf*)(base + 16 + 8 * g);
  return a.v;
}

__device__ inline v16bf load_b_frag(const bf16* Wt, int ldk, int n0, int k0, int lane) {
  // Wt is N-major [N, K]: lane col n = n0 + (lane&15); slots 0..7 -> K = k0 + 8g .. +7 ; slots 8..15 -> K = k0 + 16 + 8g .. +7
  const int n = n0 + (lane & 15), g = lane >> 4;
  const bf16* base = Wt + (long)n * ldk + k0;
  BF16x16 b;
  b.h[0] = *(const v8bf*)(base + 8 * g);
  b.h[1] = *(const v8bf*)(base + 16 + 8 * g);
  return b.v;
}

// ---------------------------------------------------------------------------
// 1) prep: weight convert/transpose to bf16, zero state + barrier.
// ---------------------------------------------------------------------------
__global__ void prep_kernel(const float* __restrict__ W_enc, const float* __restrict__ Wg,
                            const float* __restrict__ Wi,    const float* __restrict__ W_out,
                            bf16* __restrict__ W_encT, bf16* __restrict__ WgT,
                            bf16* __restrict__ WiT,    bf16* __restrict__ W_outT,
                            float* __restrict__ statesG, unsigned* __restrict__ bar) {
  const long stride = (long)gridDim.x * blockDim.x;
  const long t0     = (long)blockIdx.x * blockDim.x + threadIdx.x;

  for (long t = t0; t < (long)DM * DM; t += stride) {          // W_encT[n][k]
    long n = t >> 10, k = t & (DM - 1);
    W_encT[(n << 10) + k] = (bf16)W_enc[(k << 10) + n];
  }
  for (long t = t0; t < (long)Lnum * DM * DS; t += stride) {   // WgT[j][n][k], Wg[j][k(DS)][n(DM)]
    long j = t / ((long)DM * DS), r = t % ((long)DM * DS);
    long n = r >> 9, k = r & (DS - 1);
    WgT[t] = (bf16)Wg[j * (long)DS * DM + k * DM + n];
  }
  for (long t = t0; t < (long)Lnum * DS * DM; t += stride) {   // WiT[j][n][k], Wi[j][k(DM)][n(DS)]
    long j = t / ((long)DS * DM), r = t % ((long)DS * DM);
    long n = r >> 10, k = r & (DM - 1);
    WiT[t] = (bf16)Wi[j * (long)DM * DS + k * DS + n];
  }
  for (long t = t0; t < (long)Vocab * DS; t += stride) {       // W_outT[n][k]
    long n = t >> 9, k = t & (DS - 1);
    W_outT[t] = (bf16)W_out[k * (long)Vocab + n];
  }
  for (long t = t0; t < (long)Lnum * Bsz * DS; t += stride) statesG[t] = 0.f;
  if (t0 < 64) bar[t0] = 0u;
}

// ---------------------------------------------------------------------------
// 2) encoder: gathered GEMM, M=2048 N=1024 K=1024, bf16 WMMA.
//    grid(128, 4), block 256: 8 waves x 2 N-tiles => 16 M x 256 N per block;
//    the f32->bf16 converted A fragment feeds 2 WMMAs.
// ---------------------------------------------------------------------------
__global__ void __launch_bounds__(256) enc_gemm(const int* __restrict__ ids,
                                                const float* __restrict__ emb,
                                                const bf16* __restrict__ W_encT,
                                                const float* __restrict__ b_enc,
                                                float* __restrict__ enc_pre) {
  const int lane = threadIdx.x & 31, wave = threadIdx.x >> 5;
  const int m0 = blockIdx.x * 16;
  const int n0 = blockIdx.y * 256 + wave * 32;   // two tiles: n0, n0+16
  const int g  = lane >> 4;
  const float* arow = emb + (long)ids[m0 + (lane & 15)] * DM;  // gathered row

  v8f acc0 = {0.f, 0.f, 0.f, 0.f, 0.f, 0.f, 0.f, 0.f};
  v8f acc1 = {0.f, 0.f, 0.f, 0.f, 0.f, 0.f, 0.f, 0.f};
  for (int k0 = 0; k0 < DM; k0 += 32) {
    v8f f0 = *(const v8f*)(arow + k0 + 8 * g);
    v8f f1 = *(const v8f*)(arow + k0 + 16 + 8 * g);
    BF16x16 a;
#pragma unroll
    for (int i = 0; i < 8; ++i) { a.h[0][i] = (bf16)f0[i]; a.h[1][i] = (bf16)f1[i]; }
    v16bf b0 = load_b_frag(W_encT, DM, n0,      k0, lane);
    v16bf b1 = load_b_frag(W_encT, DM, n0 + 16, k0, lane);
    if (k0 + 32 < DM)
      __builtin_prefetch(W_encT + (long)(n0 + (lane & 15)) * DM + k0 + 32, 0, 3);
    acc0 = WMMA_BF16(a.v, b0, acc0);
    acc1 = WMMA_BF16(a.v, b1, acc1);
  }
#pragma unroll
  for (int nt = 0; nt < 2; ++nt) {
    const int n = n0 + nt * 16 + (lane & 15);
    const float bias = b_enc[n];
    const v8f acc = nt ? acc1 : acc0;
#pragma unroll
    for (int r = 0; r < 8; ++r)
      enc_pre[((long)(m0 + r + 8 * g) << 10) + n] = acc[r] + bias;
  }
}

// ---------------------------------------------------------------------------
// 3) persistent scan kernel: 16 blocks x 128 threads (4 waves).
// ---------------------------------------------------------------------------
__device__ inline void grid_sync(unsigned* bar, unsigned& phase) {
  __threadfence();                 // release this WGP's global writes (device scope)
  __syncthreads();
  if (threadIdx.x == 0) {
    const unsigned tgt = (++phase) * (unsigned)NBLK;
    __hip_atomic_fetch_add(bar, 1u, __ATOMIC_RELEASE, __HIP_MEMORY_SCOPE_AGENT);
    while (__hip_atomic_load(bar, __ATOMIC_ACQUIRE, __HIP_MEMORY_SCOPE_AGENT) < tgt)
      __builtin_amdgcn_s_sleep(1);
  }
  __syncthreads();
  __threadfence();                 // acquire before reading other WGPs' data
}

__device__ inline void layernorm_to_lds(const float* __restrict__ src, int D,
                                        const float* __restrict__ gam,
                                        const float* __restrict__ bet,
                                        bf16* __restrict__ sA, int lane, int wave) {
  for (int row = wave; row < 16; row += 4) {
    float sum = 0.f, sq = 0.f;
    for (int k = lane; k < D; k += 32) { float x = src[row * D + k]; sum += x; sq += x * x; }
#pragma unroll
    for (int off = 16; off > 0; off >>= 1) {
      sum += __shfl_xor(sum, off, 32);
      sq  += __shfl_xor(sq,  off, 32);
    }
    const float mean = sum / (float)D;
    const float var  = sq / (float)D - mean * mean;
    const float rstd = rsqrtf(var + 1e-5f);
    for (int k = lane; k < D; k += 32) {
      float x = src[row * D + k];
      sA[row * D + k] = (bf16)((x - mean) * rstd * gam[k] + bet[k]);
    }
  }
}

__global__ void __launch_bounds__(128) scan_kernel(
    const float* __restrict__ enc_pre,                    // [2048][DM]
    const bf16*  __restrict__ WgT,                        // [L][DM][DS] N-major
    const bf16*  __restrict__ WiT,                        // [L][DS][DM] N-major
    const float* __restrict__ ln_s_g, const float* __restrict__ ln_s_b,
    const float* __restrict__ ln_e_g, const float* __restrict__ ln_e_b,
    const float* __restrict__ bg,     const float* __restrict__ bi,
    float* __restrict__ statesG,                          // [L][16][DS]
    float* __restrict__ errG,                             // [16][DM]
    bf16*  __restrict__ hs,                               // [2048][DS]
    unsigned* __restrict__ bar) {
  __shared__ float s_mem_enc[16 * 64];
  __shared__ float s_mem_gen[2][16 * 64];
  __shared__ float s_mem_inf[2][16 * 32];
  __shared__ float s_bottom[16 * 64];
  __shared__ float s_red[2][256];                         // inf K-split partials
  __shared__ bf16  s_A[16 * DM];                          // layernormed A staging

  const int tid  = threadIdx.x;
  const int lane = tid & 31, wave = tid >> 5;
  const int wg   = blockIdx.x;        // 0..15, fixed column-slice owner
  const int n0g  = wg * 64;           // D_MODEL slice base
  const int n0i  = wg * 32;           // D_STATE slice base

  for (int i = tid; i < 16 * 64; i += 128) {
    s_mem_enc[i] = 0.f; s_mem_gen[0][i] = 0.f; s_mem_gen[1][i] = 0.f;
  }
  for (int i = tid; i < 16 * 32; i += 128) { s_mem_inf[0][i] = 0.f; s_mem_inf[1][i] = 0.f; }
  __syncthreads();

  unsigned phase = 0;

  for (int s = 0; s < Sseq; ++s) {
    // ---- input-encoder LIF on owned 64-column slice -> bottom spikes ----
    for (int i = tid; i < 16 * 64; i += 128) {
      const int m = i >> 6, c = i & 63;
      const float x   = enc_pre[((long)(m * Sseq + s) << 10) + n0g + c];
      const float mem = s_mem_enc[i] * DECAYF + x;
      const float sp  = (mem - THRF) >= 0.f ? 1.f : 0.f;
      s_mem_enc[i] = mem * (1.f - sp);
      s_bottom[i]  = sp;
    }
    __syncthreads();

    for (int t = 0; t < Tnum; ++t)
      for (int j = 0; j < Lnum; ++j) {
        // ---- layernorm(states[j]) -> s_A (bf16, [16][DS]) ----
        layernorm_to_lds(statesG + j * Bsz * DS, DS,
                         ln_s_g + j * DS, ln_s_b + j * DS, s_A, lane, wave);
        __syncthreads();

        // ---- generative GEMM: own [16 x 64] slice, one 16x16 tile / wave ----
        const bf16* Wg_j = WgT + (long)j * DM * DS;
        v8f acc = {0.f, 0.f, 0.f, 0.f, 0.f, 0.f, 0.f, 0.f};
        const int tn = n0g + wave * 16;
        for (int k0 = 0; k0 < DS; k0 += 32) {
          v16bf a = load_a_frag(s_A, DS, k0, lane);
          v16bf b = load_b_frag(Wg_j, DS, tn, k0, lane);
          if (k0 + 32 < DS)
            __builtin_prefetch(Wg_j + (long)(tn + (lane & 15)) * DS + k0 + 32, 0, 3);
          acc = WMMA_BF16(a, b, acc);
        }
        // ---- gen LIF + error (accumulator layout: M = r + 8g, N = lane&15) ----
        {
          const int g  = lane >> 4;
          const int nl = (lane & 15) + wave * 16;         // column within 64-slice
#pragma unroll
          for (int r = 0; r < 8; ++r) {
            const int m  = r + 8 * g;
            const int mi = m * 64 + nl;
            const float pre = acc[r] + bg[j * DM + n0g + nl];
            const float mem = s_mem_gen[j][mi] * DECAYF + pre;
            const float sp  = (mem - THRF) >= 0.f ? 1.f : 0.f;
            s_mem_gen[j][mi] = mem * (1.f - sp);
            const float e = s_bottom[mi] - sp;
            s_bottom[mi]  = e;                            // bottom <- error
            errG[(m << 10) + n0g + nl] = e;               // publish slice
          }
        }
        grid_sync(bar, phase);                            // barrier A: error visible

        // ---- layernorm(error) -> s_A (bf16, [16][DM]) ----
        layernorm_to_lds(errG, DM, ln_e_g + j * DM, ln_e_b + j * DM, s_A, lane, wave);
        __syncthreads();

        // ---- inference GEMM: own [16 x 32] slice; K split across 4 waves ----
        {
          const bf16* Wi_j = WiT + (long)j * DS * DM;
          const int tile  = wave & 1;                     // which 16x16 tile
          const int khalf = wave >> 1;                    // which K half
          v8f ac2 = {0.f, 0.f, 0.f, 0.f, 0.f, 0.f, 0.f, 0.f};
          const int ti = n0i + tile * 16;
          const int kb = khalf * (DM / 2);
          for (int k0 = kb; k0 < kb + DM / 2; k0 += 32) {
            v16bf a = load_a_frag(s_A, DM, k0, lane);
            v16bf b = load_b_frag(Wi_j, DM, ti, k0, lane);
            if (k0 + 32 < kb + DM / 2)
              __builtin_prefetch(Wi_j + (long)(ti + (lane & 15)) * DM + k0 + 32, 0, 3);
            ac2 = WMMA_BF16(a, b, ac2);
          }
          if (khalf == 1) {
#pragma unroll
            for (int r = 0; r < 8; ++r) s_red[tile][lane * 8 + r] = ac2[r];
          }
          __syncthreads();
          if (khalf == 0) {
            const int g  = lane >> 4;
            const int nl = (lane & 15) + tile * 16;       // column within 32-slice
#pragma unroll
            for (int r = 0; r < 8; ++r) {
              const int m  = r + 8 * g;
              const int mi = m * 32 + nl;
              const float pre = ac2[r] + s_red[tile][lane * 8 + r] + bi[j * DS + n0i + nl];
              const float mem = s_mem_inf[j][mi] * DECAYF + pre;
              const float sp  = (mem - THRF) >= 0.f ? 1.f : 0.f;
              s_mem_inf[j][mi] = mem * (1.f - sp);
              statesG[j * Bsz * DS + m * DS + n0i + nl] += sp;   // owner-only RMW
            }
          }
        }
        grid_sync(bar, phase);                            // barrier B: states visible
      }

    // ---- emit hs[:, s, :] = states[L-1] (own slice, written by this WGP) ----
    for (int i = tid; i < 16 * 32; i += 128) {
      const int m = i >> 5, c = i & 31;
      hs[((long)(m * Sseq + s) << 9) + n0i + c] =
          (bf16)statesG[(Lnum - 1) * Bsz * DS + m * DS + n0i + c];
    }
    __syncthreads();
  }
}

// ---------------------------------------------------------------------------
// 4) output GEMM: M=2048 N=32000 K=512 -> f32 logits + bias.
//    grid(32, 125), block 256: 64(M) x 256(N) block tile; each wave owns one
//    M-tile row and 8 N-tiles, so one A fragment feeds 8 WMMAs per k-step.
// ---------------------------------------------------------------------------
__global__ void __launch_bounds__(256) out_gemm(const bf16* __restrict__ hs,
                                                const bf16* __restrict__ W_outT,
                                                const float* __restrict__ b_out,
                                                float* __restrict__ out) {
  const int lane = threadIdx.x & 31, wave = threadIdx.x >> 5;
  const int wm = wave >> 1;                       // M-tile row 0..3
  const int wn = wave & 1;                        // N half 0..1
  const int m0 = blockIdx.x * 64 + wm * 16;
  const int nbase = blockIdx.y * 256 + wn * 128;  // 8 tiles from here
  const bf16* A = hs + (long)m0 * DS;

  v8f acc[8];
  const v8f vz = {0.f, 0.f, 0.f, 0.f, 0.f, 0.f, 0.f, 0.f};
#pragma unroll
  for (int i = 0; i < 8; ++i) acc[i] = vz;

  for (int k0 = 0; k0 < DS; k0 += 32) {
    v16bf a = load_a_frag(A, DS, k0, lane);
#pragma unroll
    for (int nt = 0; nt < 8; ++nt) {
      v16bf b = load_b_frag(W_outT, DS, nbase + nt * 16, k0, lane);
      acc[nt] = WMMA_BF16(a, b, acc[nt]);
    }
    if (k0 + 32 < DS)
      __builtin_prefetch(W_outT + (long)(nbase + (lane & 15)) * DS + k0 + 32, 0, 3);
  }

  const int g = lane >> 4;
#pragma unroll
  for (int nt = 0; nt < 8; ++nt) {
    const int n = nbase + nt * 16 + (lane & 15);
    const float bias = b_out[n];
#pragma unroll
    for (int r = 0; r < 8; ++r)
      out[(long)(m0 + r + 8 * g) * Vocab + n] = acc[nt][r] + bias;
  }
}

// ---------------------------------------------------------------------------
extern "C" void kernel_launch(void* const* d_in, const int* in_sizes, int n_in,
                              void* d_out, int out_size, void* d_ws, size_t ws_size,
                              hipStream_t stream) {
  const int*   ids    = (const int*)d_in[0];
  const float* emb    = (const float*)d_in[1];
  const float* W_enc  = (const float*)d_in[2];
  const float* b_enc  = (const float*)d_in[3];
  const float* ln_s_g = (const float*)d_in[4];
  const float* ln_s_b = (const float*)d_in[5];
  const float* Wg     = (const float*)d_in[6];
  const float* bg     = (const float*)d_in[7];
  const float* ln_e_g = (const float*)d_in[8];
  const float* ln_e_b = (const float*)d_in[9];
  const float* Wi     = (const float*)d_in[10];
  const float* bi     = (const float*)d_in[11];
  const float* W_out  = (const float*)d_in[12];
  const float* b_out  = (const float*)d_in[13];
  float*       out    = (float*)d_out;

  char* ws = (char*)d_ws;
  bf16*     W_encT  = (bf16*)(ws + OFF_WENCT);
  bf16*     WgT     = (bf16*)(ws + OFF_WGT);
  bf16*     WiT     = (bf16*)(ws + OFF_WIT);
  bf16*     W_outT  = (bf16*)(ws + OFF_WOUTT);
  float*    enc_pre = (float*)(ws + OFF_ENCP);
  bf16*     hs      = (bf16*)(ws + OFF_HS);
  float*    statesG = (float*)(ws + OFF_ST);
  float*    errG    = (float*)(ws + OFF_ERR);
  unsigned* bar     = (unsigned*)(ws + OFF_BAR);

  prep_kernel<<<2048, 256, 0, stream>>>(W_enc, Wg, Wi, W_out,
                                        W_encT, WgT, WiT, W_outT, statesG, bar);

  enc_gemm<<<dim3(NTOK / 16, DM / 256), 256, 0, stream>>>(ids, emb, W_encT, b_enc, enc_pre);

  scan_kernel<<<NBLK, 128, 0, stream>>>(enc_pre, WgT, WiT, ln_s_g, ln_s_b,
                                        ln_e_g, ln_e_b, bg, bi,
                                        statesG, errG, hs, bar);

  out_gemm<<<dim3(NTOK / 64, Vocab / 256), 256, 0, stream>>>(hs, W_outT, b_out, out);
}
